// AxialAttention_65867618452004
// MI455X (gfx1250) — matrix operand order repaired and checked by reading, not compiled, not run on hardware
//
#include <hip/hip_runtime.h>
#include <hip/hip_bf16.h>

typedef __bf16 bf16_t;
typedef __attribute__((ext_vector_type(16))) __bf16 v16bf;
typedef __attribute__((ext_vector_type(8)))  float  v8f;

constexpr int C_IN  = 512;
constexpr int C_QKV = 1536;
constexpr int NB    = 4;
constexpr int HH    = 128;
constexpr int WW    = 128;
constexpr int NPIX  = HH * WW;   // 16384
constexpr int NH    = 8;
constexpr int HD    = 64;        // head dim

// ---- WMMA helpers (this code's contiguous-k fragment layouts, wave32) ----
// A 16x32 bf16: lane m(0..15) = row M; lo lanes hold K[0..15],
// hi lanes K[16..31] -> two 16B chunks from a K-contiguous row.
__device__ __forceinline__ v16bf load_afrag(const bf16_t* row, int khalf) {
    union { uint4 q[2]; v16bf v; } u;
    u.q[0] = *(const uint4*)(row + 2 * khalf);
    u.q[1] = *(const uint4*)(row + 2 * khalf + 8);
    return u.v;
}
// B 32x16 bf16: lane = column N; K contiguous per lane -> one 32B chunk.
__device__ __forceinline__ v16bf load_bfrag(const bf16_t* colk) {
    return *(const v16bf*)colk;
}
__device__ __forceinline__ v8f wmma_bf16(v16bf a, v16bf b, v8f c) {
    return __builtin_amdgcn_wmma_f32_16x16x32_bf16(false, a, false, b, (short)0, c, false, false);
}

// ---- Tensor Data Mover (6-arg builtin on this toolchain:
//      (u32x4 g0, i32x8 g1, i32x4 g2, i32x4 g3, i32x8 extra, i32 cpol)) ----
#if __has_builtin(__builtin_amdgcn_tensor_load_to_lds)
#define HAVE_TDM 1
typedef unsigned int u32x4 __attribute__((ext_vector_type(4)));
typedef int          i32x4 __attribute__((ext_vector_type(4)));
typedef int          i32x8 __attribute__((ext_vector_type(8)));

// D# per CDNA5 ISA. Element size = 2B (bf16).
// addr = global + 2*(x + y*stride0 + z*stride1); LDS filled Z,Y,X-major.
__device__ __forceinline__ void tdm_load_tile(
    const void* gsrc, void* lds_dst,
    unsigned tile_x, unsigned tile_y, unsigned tile_z,
    unsigned long long stride0, unsigned long long stride1)
{
    unsigned long long ga = (unsigned long long)(uintptr_t)gsrc;
    unsigned lds = (unsigned)(uintptr_t)lds_dst;   // flat LDS addr low 32 = LDS offset

    u32x4 g0;
    g0.x = 1u;                                       // count=1, user descriptor
    g0.y = lds;                                      // lds_addr
    g0.z = (unsigned)ga;                             // global_addr[31:0]
    g0.w = (unsigned)((ga >> 32) & 0x01FFFFFFull) | (2u << 30); // addr[56:32], type=2

    const unsigned td0 = 0x7FFFFFFFu, td1 = 0x7FFFFFFFu;        // huge dims: no OOB clip
    i32x8 g1;
    g1[0] = (int)(1u << 16);                         // wg_mask=0, data_size=1 (2B)
    g1[1] = (int)((td0 & 0xFFFFu) << 16);            // tensor_dim0 lo
    g1[2] = (int)((td0 >> 16) | ((td1 & 0xFFFFu) << 16));
    g1[3] = (int)((td1 >> 16) | (tile_x << 16));     // tensor_dim1 hi | tile_dim0
    g1[4] = (int)(tile_y | (tile_z << 16));          // tile_dim1 | tile_dim2
    g1[5] = (int)(stride0 & 0xFFFFFFFFull);          // tensor_dim0_stride[31:0]
    g1[6] = (int)(((stride0 >> 32) & 0xFFFFull) | ((stride1 & 0xFFFFull) << 16));
    g1[7] = (int)((stride1 >> 16) & 0xFFFFFFFFull);  // tensor_dim1_stride[47:16]

    i32x4 g2; g2[0] = 0x7FFFFFFF; g2[1] = 0x7FFFFFFF; g2[2] = 0; g2[3] = 0;
    i32x4 g3; g3[0] = 0; g3[1] = 0x7FFFFFFF; g3[2] = 0; g3[3] = 0;
    i32x8 g4 = {0, 0, 0, 0, 0, 0, 0, 0};             // unused descriptor words
    __builtin_amdgcn_tensor_load_to_lds(g0, g1, g2, g3, g4, 0);
}
#endif

// =====================================================================
// Kernel 1: QKV projection, orientation M=pixels, N=out-channels so the
// epilogue writes 8 consecutive pixels per lane (one b128 store).
//   A[m=p][k=c] = x[c][p] (transposed tile), B[k=c][n=o] = Wqkv[o][c]
// Double-buffered LDS K-tiles; global loads issued before WMMA burst.
// =====================================================================
__global__ __launch_bounds__(256) void qkv_gemm_kernel(
    const float* __restrict__ x, const float* __restrict__ Wqkv,
    const float* __restrict__ bqkv, bf16_t* __restrict__ qkv)
{
    __shared__ bf16_t Xs[2][128][32];   // [pixel][c] (K contiguous)
    __shared__ bf16_t Ws[2][128][32];   // [ochan][c] (K contiguous)

    const int tid = threadIdx.x;
    const int n0  = blockIdx.x * 128;   // pixel tile
    const int m0  = blockIdx.y * 128;   // out-channel tile
    const int b   = blockIdx.z;
    const float* xb = x + (size_t)b * C_IN * NPIX;

    const int wv    = tid >> 5;
    const int lane  = tid & 31;
    const int nlane = lane & 15;
    const int khalf = (lane >> 4) << 3;
    const int kb16  = (lane >> 4) << 4;
    const int mhi   = (lane >> 4) << 3;

    // staging assignments
    const int rW  = tid >> 1;            // Ws row (o)
    const int cbW = (tid & 1) * 16;
    const int cX  = tid >> 3;            // x row (c) 0..31
    const int pbX = (tid & 7) * 16;
    const float* Wsrc = Wqkv + (size_t)(m0 + rW) * C_IN + cbW;
    const float* Xsrc = xb + (size_t)cX * NPIX + n0 + pbX;

    float4 wr[4], xr[4];
    auto ld_tile = [&](int k0) {
        #pragma unroll
        for (int j = 0; j < 4; j++) wr[j] = *(const float4*)(Wsrc + k0 + j * 4);
        #pragma unroll
        for (int j = 0; j < 4; j++) xr[j] = *(const float4*)(Xsrc + (size_t)k0 * NPIX + j * 4);
    };
    auto st_tile = [&](int buf) {
        bf16_t* dw = &Ws[buf][rW][cbW];
        #pragma unroll
        for (int j = 0; j < 4; j++) {
            dw[4*j+0] = (bf16_t)wr[j].x; dw[4*j+1] = (bf16_t)wr[j].y;
            dw[4*j+2] = (bf16_t)wr[j].z; dw[4*j+3] = (bf16_t)wr[j].w;
        }
        #pragma unroll
        for (int j = 0; j < 4; j++) {
            Xs[buf][pbX + 4*j + 0][cX] = (bf16_t)xr[j].x;
            Xs[buf][pbX + 4*j + 1][cX] = (bf16_t)xr[j].y;
            Xs[buf][pbX + 4*j + 2][cX] = (bf16_t)xr[j].z;
            Xs[buf][pbX + 4*j + 3][cX] = (bf16_t)xr[j].w;
        }
    };

    v8f acc[8] = {};
    ld_tile(0); st_tile(0);
    __syncthreads();

    for (int kt = 0; kt < C_IN / 32; kt++) {
        const int cur = kt & 1;
        if (kt + 1 < C_IN / 32) ld_tile((kt + 1) * 32);   // issue next-tile loads early
        v16bf a = load_afrag(&Xs[cur][wv * 16 + nlane][0], khalf);
        #pragma unroll
        for (int g = 0; g < 2; g++) {
            v16bf bb[4];
            #pragma unroll
            for (int s = 0; s < 4; s++) bb[s] = load_bfrag(&Ws[cur][(g*4+s)*16 + nlane][kb16]);
            #pragma unroll
            for (int s = 0; s < 4; s++) acc[g*4+s] = wmma_bf16(a, bb[s], acc[g*4+s]);
        }
        if (kt + 1 < C_IN / 32) st_tile(cur ^ 1);
        __syncthreads();
    }

    // Epilogue: lane holds 8 consecutive pixels for a fixed out-channel.
    const int p0 = n0 + wv * 16 + mhi;
    bf16_t* outb = qkv + (size_t)b * C_QKV * NPIX;
    #pragma unroll
    for (int s = 0; s < 8; s++) {
        const int o = m0 + s * 16 + nlane;
        const float bias = bqkv[o];
        union { uint4 q; bf16_t h[8]; } st;
        #pragma unroll
        for (int r = 0; r < 8; r++) st.h[r] = (bf16_t)(acc[s][r] + bias);
        *(uint4*)(outb + (size_t)o * NPIX + p0) = st.q;
    }
}

// =====================================================================
// Kernels 2/3: axial attention, ROW (along w) / COL (along h).
// V tile staged by the Tensor Data Mover while the waves transpose Q/K.
// =====================================================================
template<bool ROW>
__global__ __launch_bounds__(256) void axial_attn_kernel(
    const bf16_t* __restrict__ qkv, bf16_t* __restrict__ attn)
{
    extern __shared__ char smem[];
    bf16_t* Qs = (bf16_t*)smem;            // [128 i][64 d]
    bf16_t* Ks = Qs + 128 * 64;            // [128 i][64 d]
    bf16_t* Vs = Ks + 128 * 64;            // [64 d][128 i]
    float*  Ss = (float*)(Vs + 64 * 128);  // [128][128] f32
    bf16_t* Ps = (bf16_t*)(Ss + 128 * 128);// [128][128] bf16

    const int tid  = threadIdx.x;
    const int line = blockIdx.x;           // h for ROW, w for COL
    const int head = blockIdx.y;
    const int b    = blockIdx.z;

    const bf16_t* qb = qkv + ((size_t)b * C_QKV + head * HD) * NPIX;
    const bf16_t* kb = qb + (size_t)C_IN * NPIX;
    const bf16_t* vb = qb + (size_t)(2 * C_IN) * NPIX;
    const size_t  boff    = ROW ? (size_t)line * WW : (size_t)line;
    const int     istride = ROW ? 1 : WW;

#if defined(HAVE_TDM)
    if (tid < 32) {                         // wave 0 drives the TDM (EXEC-independent)
        if (ROW) tdm_load_tile(vb + boff, Vs, 128, 64, 0,  (unsigned long long)NPIX, 0ull);
        else     tdm_load_tile(vb + boff, Vs, 1, 128, 64,  (unsigned long long)WW,
                               (unsigned long long)NPIX);   // element gather -> [d][h]
    }
#endif

    { // Stage Q^T, K^T (transpose into [i][d]); V too if no TDM.
        const int d  = tid >> 2;
        const int ib = (tid & 3) * 32;
        const bf16_t* qs = qb + (size_t)d * NPIX + boff;
        const bf16_t* ks = kb + (size_t)d * NPIX + boff;
        if (ROW) {
            union { uint4 q[4]; bf16_t h[32]; } qu, ku;
            #pragma unroll
            for (int j = 0; j < 4; j++) {
                qu.q[j] = *(const uint4*)(qs + ib + j * 8);
                ku.q[j] = *(const uint4*)(ks + ib + j * 8);
            }
            #pragma unroll
            for (int j = 0; j < 32; j++) {
                Qs[(ib + j) * 64 + d] = qu.h[j];
                Ks[(ib + j) * 64 + d] = ku.h[j];
            }
        } else {
            #pragma unroll 8
            for (int j = 0; j < 32; j++) {
                const size_t o = (size_t)(ib + j) * istride;
                Qs[(ib + j) * 64 + d] = qs[o];
                Ks[(ib + j) * 64 + d] = ks[o];
            }
        }
#if !defined(HAVE_TDM)
        const bf16_t* vs = vb + (size_t)d * NPIX + boff;
        #pragma unroll 8
        for (int j = 0; j < 32; j++)
            Vs[d * 128 + ib + j] = vs[(size_t)(ib + j) * istride];
#endif
    }
#if defined(HAVE_TDM)
    if (tid < 32) __builtin_amdgcn_s_wait_tensorcnt(0);
#endif
    __syncthreads();

    const int wv    = tid >> 5;
    const int lane  = tid & 31;
    const int nlane = lane & 15;
    const int khalf = (lane >> 4) << 3;
    const int kb16  = (lane >> 4) << 4;
    const int mhi   = (lane >> 4) << 3;
    const float scale = 0.125f;            // 1/sqrt(64)

    // ---- S = Q^T K : M=128(i), N=128(u), K=64(d) ----
    v8f acc[8] = {};
    #pragma unroll
    for (int k0 = 0; k0 < 64; k0 += 32) {
        v16bf a = load_afrag(Qs + (wv * 16 + nlane) * 64 + k0, khalf);
        #pragma unroll
        for (int g = 0; g < 2; g++) {
            v16bf bb[4];
            #pragma unroll
            for (int s = 0; s < 4; s++)
                bb[s] = load_bfrag(Ks + ((g*4+s) * 16 + nlane) * 64 + k0 + kb16);
            #pragma unroll
            for (int s = 0; s < 4; s++) acc[g*4+s] = wmma_bf16(a, bb[s], acc[g*4+s]);
        }
    }
    #pragma unroll
    for (int s = 0; s < 8; s++) {
        const int u = s * 16 + nlane;
        #pragma unroll
        for (int r = 0; r < 8; r++)
            Ss[(wv * 16 + mhi + r) * 128 + u] = acc[s][r] * scale;
    }
    __syncthreads();

    // ---- softmax: 2 threads per row, shfl_xor to combine halves ----
    {
        const int row  = tid >> 1;
        const int half = (tid & 1) * 64;
        float* rp = Ss + row * 128 + half;
        float m = rp[0];
        #pragma unroll 8
        for (int u = 1; u < 64; u++) m = fmaxf(m, rp[u]);
        m = fmaxf(m, __shfl_xor(m, 1, 32));
        float sum = 0.f;
        #pragma unroll 8
        for (int u = 0; u < 64; u++) { float e = __expf(rp[u] - m); rp[u] = e; sum += e; }
        sum += __shfl_xor(sum, 1, 32);
        const float inv = 1.f / sum;
        bf16_t* pp = Ps + row * 128 + half;
        #pragma unroll 8
        for (int u = 0; u < 64; u++) pp[u] = (bf16_t)(rp[u] * inv);
    }
    __syncthreads();

    // ---- out^T[i][d] = P[i][:] . V^T : M=128(i), N=64(d), K=128(u) ----
    v8f acc2[4] = {};
    #pragma unroll
    for (int k0 = 0; k0 < 128; k0 += 32) {
        v16bf a = load_afrag(Ps + (wv * 16 + nlane) * 128 + k0, khalf);
        v16bf bb[4];
        #pragma unroll
        for (int s = 0; s < 4; s++)
            bb[s] = load_bfrag(Vs + (s * 16 + nlane) * 128 + k0 + kb16);
        #pragma unroll
        for (int s = 0; s < 4; s++) acc2[s] = wmma_bf16(a, bb[s], acc2[s]);
    }

    bf16_t* ab = attn + ((size_t)b * C_IN + head * HD) * NPIX;
    const int i0 = wv * 16 + mhi;          // 8 consecutive positions along axis
    #pragma unroll
    for (int s = 0; s < 4; s++) {
        const int d = s * 16 + nlane;
        if (ROW) {                          // contiguous: one b128 store
            union { uint4 q; bf16_t h[8]; } st;
            #pragma unroll
            for (int r = 0; r < 8; r++) st.h[r] = (bf16_t)acc2[s][r];
            *(uint4*)(ab + (size_t)d * NPIX + boff + i0) = st.q;
        } else {                            // strided RMW accumulate
            #pragma unroll
            for (int r = 0; r < 8; r++) {
                const size_t off = (size_t)d * NPIX + boff + (size_t)(i0 + r) * istride;
                ab[off] = (bf16_t)((float)ab[off] + acc2[s][r]);
            }
        }
    }
}

// =====================================================================
// Kernel 4: output projection (same swapped orientation, fp32 out).
// =====================================================================
__global__ __launch_bounds__(256) void out_proj_kernel(
    const bf16_t* __restrict__ attn, const float* __restrict__ Wout,
    const float* __restrict__ bout, float* __restrict__ out)
{
    __shared__ bf16_t Xs[2][128][32];   // [pixel][c]
    __shared__ bf16_t Ws[2][128][32];   // [ochan][c]

    const int tid = threadIdx.x;
    const int n0  = blockIdx.x * 128;
    const int m0  = blockIdx.y * 128;
    const int b   = blockIdx.z;
    const bf16_t* ab = attn + (size_t)b * C_IN * NPIX;

    const int wv    = tid >> 5;
    const int lane  = tid & 31;
    const int nlane = lane & 15;
    const int khalf = (lane >> 4) << 3;
    const int kb16  = (lane >> 4) << 4;
    const int mhi   = (lane >> 4) << 3;

    const int rW  = tid >> 1;
    const int cbW = (tid & 1) * 16;
    const int cX  = tid >> 3;
    const int pbX = (tid & 7) * 16;
    const float*  Wsrc = Wout + (size_t)(m0 + rW) * C_IN + cbW;
    const bf16_t* Xsrc = ab + (size_t)cX * NPIX + n0 + pbX;

    float4 wr[4];
    union { uint4 q[2]; bf16_t h[16]; } xr;
    auto ld_tile = [&](int k0) {
        #pragma unroll
        for (int j = 0; j < 4; j++) wr[j] = *(const float4*)(Wsrc + k0 + j * 4);
        xr.q[0] = *(const uint4*)(Xsrc + (size_t)k0 * NPIX);
        xr.q[1] = *(const uint4*)(Xsrc + (size_t)k0 * NPIX + 8);
    };
    auto st_tile = [&](int buf) {
        bf16_t* dw = &Ws[buf][rW][cbW];
        #pragma unroll
        for (int j = 0; j < 4; j++) {
            dw[4*j+0] = (bf16_t)wr[j].x; dw[4*j+1] = (bf16_t)wr[j].y;
            dw[4*j+2] = (bf16_t)wr[j].z; dw[4*j+3] = (bf16_t)wr[j].w;
        }
        #pragma unroll
        for (int j = 0; j < 16; j++) Xs[buf][pbX + j][cX] = xr.h[j];
    };

    v8f acc[8] = {};
    ld_tile(0); st_tile(0);
    __syncthreads();

    for (int kt = 0; kt < C_IN / 32; kt++) {
        const int cur = kt & 1;
        if (kt + 1 < C_IN / 32) ld_tile((kt + 1) * 32);
        v16bf a = load_afrag(&Xs[cur][wv * 16 + nlane][0], khalf);
        #pragma unroll
        for (int g = 0; g < 2; g++) {
            v16bf bb[4];
            #pragma unroll
            for (int s = 0; s < 4; s++) bb[s] = load_bfrag(&Ws[cur][(g*4+s)*16 + nlane][kb16]);
            #pragma unroll
            for (int s = 0; s < 4; s++) acc[g*4+s] = wmma_bf16(a, bb[s], acc[g*4+s]);
        }
        if (kt + 1 < C_IN / 32) st_tile(cur ^ 1);
        __syncthreads();
    }

    const int p0 = n0 + wv * 16 + mhi;     // 8 consecutive pixels
    float* outb = out + (size_t)b * C_IN * NPIX;
    #pragma unroll
    for (int s = 0; s < 8; s++) {
        const int o = m0 + s * 16 + nlane;
        const float bias = bout[o];
        union { float4 f[2]; float v[8]; } st;
        #pragma unroll
        for (int r = 0; r < 8; r++) st.v[r] = acc[s][r] + bias;
        float* dst = outb + (size_t)o * NPIX + p0;
        *(float4*)(dst + 0) = st.f[0];
        *(float4*)(dst + 4) = st.f[1];
    }
}

extern "C" void kernel_launch(void* const* d_in, const int* in_sizes, int n_in,
                              void* d_out, int out_size, void* d_ws, size_t ws_size,
                              hipStream_t stream) {
    const float* x    = (const float*)d_in[0];
    const float* Wqkv = (const float*)d_in[1];
    const float* bqkv = (const float*)d_in[2];
    const float* Wout = (const float*)d_in[3];
    const float* bout = (const float*)d_in[4];
    float* out = (float*)d_out;

    bf16_t* qkv  = (bf16_t*)d_ws;                      // 4*1536*16384 bf16 = 192 MB
    bf16_t* attn = qkv + (size_t)NB * C_QKV * NPIX;    // 4*512*16384 bf16  =  64 MB

    dim3 blk(256);
    qkv_gemm_kernel<<<dim3(NPIX / 128, C_QKV / 128, NB), blk, 0, stream>>>(x, Wqkv, bqkv, qkv);

    size_t smem = (size_t)(3 * 128 * 64) * sizeof(bf16_t)
                + (size_t)(128 * 128) * sizeof(float)
                + (size_t)(128 * 128) * sizeof(bf16_t);
    axial_attn_kernel<true ><<<dim3(HH, NH, NB), blk, smem, stream>>>(qkv, attn);
    axial_attn_kernel<false><<<dim3(WW, NH, NB), blk, smem, stream>>>(qkv, attn);

    out_proj_kernel<<<dim3(NPIX / 128, C_IN / 128, NB), blk, 0, stream>>>(attn, Wout, bout, out);
}
